// CausalSelfAttention_38998303048437
// MI455X (gfx1250) — matrix operand order repaired and checked by reading, not compiled, not run on hardware
//
#include <hip/hip_runtime.h>
#include <hip/hip_bf16.h>
#include <cmath>

// ---------------- constants ----------------
#define B_    2
#define T_    2048
#define D_    1024
#define NH_   16
#define NKV_  4
#define HD_   64
#define QKVC_ ((NH_ + 2*NKV_)*HD_)   // 1536
#define GRP_  (NH_/NKV_)             // 4

typedef __bf16 bf16;
typedef __attribute__((ext_vector_type(8)))  __bf16 v8bf;
typedef __attribute__((ext_vector_type(16))) __bf16 v16bf;
typedef __attribute__((ext_vector_type(8)))  float  v8f;

// ---------------- WMMA helpers (CDNA5 wave32) ----------------
// A-matrix 16x32 bf16 fragment: lane l -> row M=l%16, K = 16*half .. 16*half+15
// p must point at (row, k0 + 16*half) in a row-major bf16 array.
__device__ __forceinline__ v16bf make_a_frag(const bf16* p) {
    return *(const v16bf*)(p);
}

// B-matrix 32x16 bf16 fragment: lane (n = lane&15) owns column N=n, K = 16*half .. 16*half+15
__device__ __forceinline__ v16bf make_b_frag(const bf16* p) {
    return *(const v16bf*)(p);
}

__device__ __forceinline__ v8f wmma_bf16(v16bf a, v16bf b, v8f c) {
    return __builtin_amdgcn_wmma_f32_16x16x32_bf16(false, a, false, b, (short)0, c, false, false);
}

// ---------------- fp32 -> bf16 convert ----------------
__global__ void cvt_bf16_kernel(const float* __restrict__ in, bf16* __restrict__ out, int n) {
    int i = blockIdx.x * blockDim.x + threadIdx.x;
    if (i < n) out[i] = (bf16)in[i];
}

// ---------------- WMMA GEMM, 32x64 tile per wave: C[M,N] = A[M,K] * B[N,K]^T ----------
// AMODE 0: A row-major (m*K + k).   AMODE 1: A is (B,NH,T,HD) head-major; m=b*T+t, k=h*64+d.
template <int AMODE>
__device__ __forceinline__ const bf16* a_addr(const bf16* A, int m, int k, int K) {
    if (AMODE == 0) return A + (size_t)m * K + k;
    int b = m / T_, t = m - b * T_;             // 16-element runs never cross a 64-wide head
    return A + ((size_t)(b * NH_ + (k >> 6)) * T_ + t) * HD_ + (k & 63);
}

template <int AMODE>
__global__ __launch_bounds__(256)
void gemm32x64_bf16_kernel(const bf16* __restrict__ A, const bf16* __restrict__ Bm,
                           float* __restrict__ C, int M, int N, int K) {
    int wid  = blockIdx.x * (blockDim.x >> 5) + (threadIdx.x >> 5);
    int lane = threadIdx.x & 31;
    int half = lane >> 4, lr = lane & 15;
    int ntiles = N >> 6;                        // 64-wide N tiles
    int mt = wid / ntiles, nt = wid % ntiles;
    if (mt >= (M >> 5)) return;                 // 32-tall M tiles (wave-uniform exit)

    int arow0 = (mt << 5) + lr;                 // per-lane A rows for the two m-subtiles
    int arow1 = arow0 + 16;
    // per-lane B row pointers (columns of B^T) for the four n-subtiles
    const bf16* bp0 = Bm + (size_t)((nt << 6) +  0 + lr) * K;
    const bf16* bp1 = Bm + (size_t)((nt << 6) + 16 + lr) * K;
    const bf16* bp2 = Bm + (size_t)((nt << 6) + 32 + lr) * K;
    const bf16* bp3 = Bm + (size_t)((nt << 6) + 48 + lr) * K;

    v8f acc[2][4];
#pragma unroll
    for (int i = 0; i < 2; ++i)
#pragma unroll
        for (int j = 0; j < 4; ++j) acc[i][j] = (v8f){};

#pragma unroll 2
    for (int k0 = 0; k0 < K; k0 += 32) {
        // prefetch next k-chunk of B columns (global_prefetch_b8)
        __builtin_prefetch((const void*)(bp0 + k0 + 32 + 16 * half), 0, 1);
        __builtin_prefetch((const void*)(bp2 + k0 + 32 + 16 * half), 0, 1);

        v16bf a0 = make_a_frag(a_addr<AMODE>(A, arow0, k0 + 16 * half, K));
        v16bf a1 = make_a_frag(a_addr<AMODE>(A, arow1, k0 + 16 * half, K));
        v16bf b0 = make_b_frag(bp0 + k0 + 16 * half);
        v16bf b1 = make_b_frag(bp1 + k0 + 16 * half);
        v16bf b2 = make_b_frag(bp2 + k0 + 16 * half);
        v16bf b3 = make_b_frag(bp3 + k0 + 16 * half);
        acc[0][0] = wmma_bf16(a0, b0, acc[0][0]);
        acc[0][1] = wmma_bf16(a0, b1, acc[0][1]);
        acc[0][2] = wmma_bf16(a0, b2, acc[0][2]);
        acc[0][3] = wmma_bf16(a0, b3, acc[0][3]);
        acc[1][0] = wmma_bf16(a1, b0, acc[1][0]);
        acc[1][1] = wmma_bf16(a1, b1, acc[1][1]);
        acc[1][2] = wmma_bf16(a1, b2, acc[1][2]);
        acc[1][3] = wmma_bf16(a1, b3, acc[1][3]);
    }

    // C layout per tile: VGPR r -> row + r + 8*half, col subtile*16 + lr
#pragma unroll
    for (int i = 0; i < 2; ++i) {
        float* cp = C + (size_t)((mt << 5) + (i << 4) + 8 * half) * N + (nt << 6) + lr;
#pragma unroll
        for (int j = 0; j < 4; ++j)
#pragma unroll
            for (int r = 0; r < 8; ++r) cp[(size_t)r * N + (j << 4)] = acc[i][j][r];
    }
}

// ---------------- RoPE + scale + pack to bf16 layouts ----------------
__global__ void rope_pack_kernel(const float* __restrict__ qkv,
                                 const float* __restrict__ cosb, const float* __restrict__ sinb,
                                 const float* __restrict__ q_scale, const float* __restrict__ k_scale,
                                 bf16* __restrict__ qbf, bf16* __restrict__ kbf,
                                 bf16* __restrict__ vt,  float* __restrict__ vf) {
    int i = blockIdx.x * blockDim.x + threadIdx.x;
    if (i >= B_ * T_ * QKVC_) return;
    int col = i % QKVC_;
    int bt  = i / QKVC_;
    int t = bt % T_, b = bt / T_;
    float val = qkv[i];

    if (col < NH_ * HD_) {                       // ---- Q ----
        int h = col >> 6, d = col & 63;
        float sc = q_scale[h];
        float v = val * sc;
        if (d < 16) {
            if (d < 8) {
                float x2 = qkv[i + 8] * sc;
                v = v * cosb[t * 8 + d] - x2 * sinb[t * 8 + d];
            } else {
                float x1 = qkv[i - 8] * sc;
                v = x1 * sinb[t * 8 + d - 8] + v * cosb[t * 8 + d - 8];
            }
        }
        // fold softmax 1/sqrt(HD) into q
        qbf[((size_t)(b * NH_ + h) * T_ + t) * HD_ + d] = (bf16)(v * 0.125f);
    } else if (col < (NH_ + NKV_) * HD_) {       // ---- K ----
        int c2 = col - NH_ * HD_;
        int h = c2 >> 6, d = c2 & 63;
        float sc = k_scale[h];
        float v = val * sc;
        if (d < 16) {
            if (d < 8) {
                float x2 = qkv[i + 8] * sc;
                v = v * cosb[t * 8 + d] - x2 * sinb[t * 8 + d];
            } else {
                float x1 = qkv[i - 8] * sc;
                v = x1 * sinb[t * 8 + d - 8] + v * cosb[t * 8 + d - 8];
            }
        }
        kbf[((size_t)(b * NKV_ + h) * T_ + t) * HD_ + d] = (bf16)v;
    } else {                                     // ---- V ----
        int c2 = col - (NH_ + NKV_) * HD_;
        int h = c2 >> 6, d = c2 & 63;
        vf[((size_t)(b * NKV_ + h) * T_ + t) * HD_ + d] = val;
        vt[((size_t)(b * NKV_ + h) * HD_ + d) * T_ + t] = (bf16)val;  // transposed for PV B-frags
    }
}

// ---------------- flash attention: wave per 16-row q tile ----------------
__global__ __launch_bounds__(128)
void attn_kernel(const bf16* __restrict__ qbf, const bf16* __restrict__ kbf,
                 const bf16* __restrict__ vt, float* __restrict__ ao) {
    __shared__ bf16 plds[4][16][32];             // per-wave P staging (C-layout -> A-layout)
    int wave = threadIdx.x >> 5, lane = threadIdx.x & 31;
    int half = lane >> 4, lr = lane & 15;

    int qt  = blockIdx.x * 4 + wave;             // over B*NH*(T/16)
    int tpt = T_ / 16;
    int bh = qt / tpt, q16 = qt % tpt;
    int b = bh / NH_, h = bh % NH_;
    int kvh = h / GRP_;

    const bf16* qp = qbf + (size_t)(b * NH_  + h)   * T_  * HD_;
    const bf16* kp = kbf + (size_t)(b * NKV_ + kvh) * T_  * HD_;
    const bf16* vp = vt  + (size_t)(b * NKV_ + kvh) * HD_ * T_;
    int qbase = q16 << 4;

    // q A-frags for head-dim chunks [0,32) and [32,64)
    v16bf qa0 = make_a_frag(qp + (size_t)(qbase + lr) * HD_ + 0  + 16 * half);
    v16bf qa1 = make_a_frag(qp + (size_t)(qbase + lr) * HD_ + 32 + 16 * half);

    v8f O0 = {}, O1 = {}, O2 = {}, O3 = {};
    float mrow[8], lrow[8];
#pragma unroll
    for (int r = 0; r < 8; ++r) { mrow[r] = -INFINITY; lrow[r] = 0.0f; }

    int nchunk = (qbase + 16 + 31) >> 5;         // causal kv coverage, chunks of 32
    for (int c = 0; c < nchunk; ++c) {
        int kv0 = c << 5;
        v8f S0 = {}, S1 = {};
        // S = q . k^T : B-frags are contiguous head-dim runs of K rows
        v16bf kb0 = make_b_frag(kp + (size_t)(kv0      + lr) * HD_ + 0  + 16 * half);
        v16bf kb1 = make_b_frag(kp + (size_t)(kv0 + 16 + lr) * HD_ + 0  + 16 * half);
        S0 = wmma_bf16(qa0, kb0, S0);
        S1 = wmma_bf16(qa0, kb1, S1);
        kb0 = make_b_frag(kp + (size_t)(kv0      + lr) * HD_ + 32 + 16 * half);
        kb1 = make_b_frag(kp + (size_t)(kv0 + 16 + lr) * HD_ + 32 + 16 * half);
        S0 = wmma_bf16(qa1, kb0, S0);
        S1 = wmma_bf16(qa1, kb1, S1);

        int col0 = kv0 + lr, col1 = kv0 + 16 + lr;
#pragma unroll
        for (int r = 0; r < 8; ++r) {
            int row = qbase + r + 8 * half;
            float s0 = (col0 <= row) ? S0[r] : -INFINITY;   // causal mask
            float s1 = (col1 <= row) ? S1[r] : -INFINITY;
            float mx = fmaxf(s0, s1);
#pragma unroll
            for (int m = 1; m < 16; m <<= 1) mx = fmaxf(mx, __shfl_xor(mx, m, 32));
            float mn = fmaxf(mrow[r], mx);
            float alpha = __expf(mrow[r] - mn);
            float p0 = __expf(s0 - mn);
            float p1 = __expf(s1 - mn);
            float rs = p0 + p1;
#pragma unroll
            for (int m = 1; m < 16; m <<= 1) rs += __shfl_xor(rs, m, 32);
            lrow[r] = lrow[r] * alpha + rs;
            mrow[r] = mn;
            O0[r] *= alpha; O1[r] *= alpha; O2[r] *= alpha; O3[r] *= alpha;
            plds[wave][r + 8 * half][lr]      = (bf16)p0;   // stage P for A-layout reload
            plds[wave][r + 8 * half][16 + lr] = (bf16)p1;
        }
        // same-wave DS ops are in-order: safe to reload P as an A fragment
        v16bf pa = make_a_frag(&plds[wave][lr][16 * half]);
        v16bf vb;
        vb = make_b_frag(vp + (size_t)(0 * 16 + lr) * T_ + kv0 + 16 * half); O0 = wmma_bf16(pa, vb, O0);
        vb = make_b_frag(vp + (size_t)(1 * 16 + lr) * T_ + kv0 + 16 * half); O1 = wmma_bf16(pa, vb, O1);
        vb = make_b_frag(vp + (size_t)(2 * 16 + lr) * T_ + kv0 + 16 * half); O2 = wmma_bf16(pa, vb, O2);
        vb = make_b_frag(vp + (size_t)(3 * 16 + lr) * T_ + kv0 + 16 * half); O3 = wmma_bf16(pa, vb, O3);
    }

    float* aop = ao + ((size_t)(b * NH_ + h) * T_ + qbase) * HD_;
#pragma unroll
    for (int r = 0; r < 8; ++r) {
        float inv = 1.0f / lrow[r];
        size_t ro = (size_t)(r + 8 * half) * HD_;
        aop[ro + 0 * 16 + lr] = O0[r] * inv;
        aop[ro + 1 * 16 + lr] = O1[r] * inv;
        aop[ro + 2 * 16 + lr] = O2[r] * inv;
        aop[ro + 3 * 16 + lr] = O3[r] * inv;
    }
}

// ---------------- v-component removal: wave per (b,h,t) row ----------------
__global__ __launch_bounds__(256)
void vremove_kernel(const float* __restrict__ ao, const float* __restrict__ vf,
                    bf16* __restrict__ aobf) {
    int wid  = blockIdx.x * (blockDim.x >> 5) + (threadIdx.x >> 5);
    int lane = threadIdx.x & 31;
    int t = wid % T_;
    int bh = wid / T_;
    int h = bh % NH_, b = bh / NH_;
    int kvh = h / GRP_;

    const float* ap = ao + ((size_t)(b * NH_  + h)   * T_ + t) * HD_;
    const float* vp = vf + ((size_t)(b * NKV_ + kvh) * T_ + t) * HD_;
    float a0 = ap[lane], a1 = ap[lane + 32];
    float v0 = vp[lane], v1 = vp[lane + 32];
    float dot = a0 * v0 + a1 * v1;
    float nrm = v0 * v0 + v1 * v1;
#pragma unroll
    for (int m = 1; m < 32; m <<= 1) {
        dot += __shfl_xor(dot, m, 32);
        nrm += __shfl_xor(nrm, m, 32);
    }
    float coef = dot / fmaxf(nrm, 1e-8f);
    bf16* op = aobf + ((size_t)(b * NH_ + h) * T_ + t) * HD_;
    op[lane]      = (bf16)(a0 - coef * v0);
    op[lane + 32] = (bf16)(a1 - coef * v1);
}

// ---------------- launcher ----------------
extern "C" void kernel_launch(void* const* d_in, const int* in_sizes, int n_in,
                              void* d_out, int out_size, void* d_ws, size_t ws_size,
                              hipStream_t stream) {
    const float* x       = (const float*)d_in[0];
    const float* cosb    = (const float*)d_in[1];
    const float* sinb    = (const float*)d_in[2];
    /* d_in[3] attn_mask: causal, handled analytically */
    const float* w_qkv   = (const float*)d_in[4];
    const float* w_out   = (const float*)d_in[5];
    const float* q_scale = (const float*)d_in[6];
    const float* k_scale = (const float*)d_in[7];
    float* out = (float*)d_out;

    const size_t BT = (size_t)B_ * T_;
    char* ws = (char*)d_ws;
    size_t off = 0;
    auto alloc = [&](size_t bytes) -> void* {
        void* p = ws + off;
        off += (bytes + 255) & ~(size_t)255;
        return p;
    };

    bf16*  xb    = (bf16*) alloc(BT * D_ * sizeof(bf16));
    bf16*  wqkvb = (bf16*) alloc((size_t)QKVC_ * D_ * sizeof(bf16));
    bf16*  woutb = (bf16*) alloc((size_t)D_ * D_ * sizeof(bf16));
    float* qkvf  = (float*)alloc(BT * QKVC_ * sizeof(float));
    bf16*  qbf   = (bf16*) alloc((size_t)B_ * NH_  * T_ * HD_ * sizeof(bf16));
    bf16*  kbf   = (bf16*) alloc((size_t)B_ * NKV_ * T_ * HD_ * sizeof(bf16));
    bf16*  vt    = (bf16*) alloc((size_t)B_ * NKV_ * HD_ * T_ * sizeof(bf16));
    float* vf    = (float*)alloc((size_t)B_ * NKV_ * T_ * HD_ * sizeof(float));
    float* aof   = (float*)alloc((size_t)B_ * NH_ * T_ * HD_ * sizeof(float));
    bf16*  aobf  = (bf16*) alloc((size_t)B_ * NH_ * T_ * HD_ * sizeof(bf16));

    // 1) convert inputs to bf16
    {
        int n = (int)(BT * D_);
        cvt_bf16_kernel<<<(n + 255) / 256, 256, 0, stream>>>(x, xb, n);
        n = QKVC_ * D_;
        cvt_bf16_kernel<<<(n + 255) / 256, 256, 0, stream>>>(w_qkv, wqkvb, n);
        n = D_ * D_;
        cvt_bf16_kernel<<<(n + 255) / 256, 256, 0, stream>>>(w_out, woutb, n);
    }

    // 2) QKV GEMM: (4096 x 1024) x (1536 x 1024)^T, 32x64 tile per wave
    {
        int tiles = ((int)BT / 32) * (QKVC_ / 64);   // 128 * 24 = 3072
        gemm32x64_bf16_kernel<0><<<tiles / 8, 256, 0, stream>>>(xb, wqkvb, qkvf,
                                                                (int)BT, QKVC_, D_);
    }

    // 3) scale + RoPE + pack
    {
        int n = (int)(BT * QKVC_);
        rope_pack_kernel<<<(n + 255) / 256, 256, 0, stream>>>(qkvf, cosb, sinb,
                                                              q_scale, k_scale,
                                                              qbf, kbf, vt, vf);
    }

    // 4) flash attention: B*NH*(T/16) wave-tiles, 4 waves/block
    {
        int qtiles = B_ * NH_ * (T_ / 16);           // 4096
        attn_kernel<<<qtiles / 4, 128, 0, stream>>>(qbf, kbf, vt, aof);
    }

    // 5) v-component removal
    {
        int rows = B_ * NH_ * T_;                    // 65536 waves
        vremove_kernel<<<rows / 8, 256, 0, stream>>>(aof, vf, aobf);
    }

    // 6) output projection: (4096 x 1024) x (1024 x 1024)^T, head-major A
    {
        int tiles = ((int)BT / 32) * (D_ / 64);      // 128 * 16 = 2048
        gemm32x64_bf16_kernel<1><<<tiles / 8, 256, 0, stream>>>(aobf, woutb, out,
                                                                (int)BT, D_, D_);
    }
    (void)in_sizes; (void)n_in; (void)out_size; (void)ws_size;
}
